// HiGraphLatentUnet_37314675868363
// MI455X (gfx1250) — compile-verified, matrix operand order checked by reading, not run on hardware
//
#include <hip/hip_runtime.h>
#include <hip/hip_bf16.h>
#include <cstdint>
#include <cstddef>

#define HDIM 128

// ---------------------------------------------------------------------------
// CDNA5 WMMA types
// ---------------------------------------------------------------------------
typedef __attribute__((ext_vector_type(16))) __bf16 v16bf;
typedef __attribute__((ext_vector_type(8)))  float  v8f;

union Frag {
  v16bf v;
  uint4 q[2];
};

__device__ __forceinline__ unsigned pk_bf16(float a, float b) {
  unsigned ua = __float_as_uint(a);
  unsigned ub = __float_as_uint(b);
  ua = (ua + 0x7FFFu + ((ua >> 16) & 1u)) >> 16;   // round-to-nearest-even
  ub = (ub + 0x7FFFu + ((ub >> 16) & 1u)) >> 16;
  return ua | (ub << 16);
}

__device__ __forceinline__ unsigned short f2bf(float a) {
  unsigned u = __float_as_uint(a);
  u = (u + 0x7FFFu + ((u >> 16) & 1u)) >> 16;
  return (unsigned short)u;
}

// ---------------------------------------------------------------------------
// Weight converter: f32 [K x N] row-major  ->  bf16 transposed [Npad x K]
// (rows beyond N zero-padded, so WMMA B-fragments are contiguous in K)
// ---------------------------------------------------------------------------
__global__ __launch_bounds__(256) void cvt_wT_kernel(
    const float* __restrict__ W, unsigned short* __restrict__ WT,
    int K, int N, int Npad) {
  int i = blockIdx.x * 256 + threadIdx.x;
  if (i >= Npad * K) return;
  int n = i / K, k = i - n * K;
  float v = (n < N) ? W[(size_t)k * N + n] : 0.f;
  WT[(size_t)n * K + k] = f2bf(v);
}

// ---------------------------------------------------------------------------
// Fused 2-layer MLP: gather-concat -> Linear(NCAT*128 -> 128) -> SiLU
//                    -> Linear(128 -> 128) -> [LayerNorm]
//                    -> scatter-sum (pre-residual) / residual row store
// Tile: 16 rows per block, 128 threads (4 waves), 32 output cols per wave.
// All GEMMs via v_wmma_f32_16x16x32_bf16.
// ---------------------------------------------------------------------------
template <int NCAT, bool LN, bool PMAP>
__global__ __launch_bounds__(128) void fused_mlp_kernel(
    const float* __restrict__ cat0, const int* __restrict__ idx0,
    const float* __restrict__ cat1, const int* __restrict__ idx1,
    const float* __restrict__ cat2, const int* __restrict__ idx2,
    const unsigned short* __restrict__ W1T, const float* __restrict__ b1,
    const unsigned short* __restrict__ W2T, const float* __restrict__ b2,
    const float* __restrict__ lnS, const float* __restrict__ lnB,
    const float* __restrict__ resid,
    float* __restrict__ outRow,
    float* __restrict__ outScatter, const int* __restrict__ sidx,
    int nrows) {
  constexpr int KTOT = NCAT * HDIM;
  __shared__ alignas(16) unsigned short Abuf[16 * KTOT];
  __shared__ alignas(16) unsigned short Hbuf[16 * HDIM];
  __shared__ alignas(16) float          Obuf[16 * HDIM];

  const int tid = threadIdx.x;
  const int tile0 = blockIdx.x * 16;

  // ---- stage: gather + f32->bf16 into LDS (row-major [16][KTOT]) ----
#pragma unroll
  for (int it = 0; it < NCAT * 4; ++it) {
    int e4 = it * 128 + tid;                 // float4 group id
    int row = e4 / (KTOT / 4);
    int k = (e4 % (KTOT / 4)) * 4;
    int rg = tile0 + row;
    float4 x = make_float4(0.f, 0.f, 0.f, 0.f);
    if (rg < nrows) {
      int c = k >> 7;
      int kk = k & 127;
      const float* base = cat0;
      const int* ix = idx0;
      if (NCAT > 1 && c == 1) { base = cat1; ix = idx1; }
      if (NCAT > 2 && c == 2) { base = cat2; ix = idx2; }
      int r2 = ix ? ix[rg] : rg;
      x = *(const float4*)(base + (size_t)r2 * HDIM + kk);
    }
    uint2 p;
    p.x = pk_bf16(x.x, x.y);
    p.y = pk_bf16(x.z, x.w);
    *(uint2*)&Abuf[row * KTOT + k] = p;
  }
  __syncthreads();

  const int lane = tid & 31;
  const int wv = tid >> 5;
  const int rowA = lane & 15;           // A-fragment: M row
  const int kh = (lane >> 4) * 8;       // K half-group select
  const int mBase = (lane >> 4) * 8;    // C/D: M = r + mBase
  const int nA = lane & 15;             // C/D & B: N column within tile
  const int n0 = wv * 32;
  const int n1 = n0 + 16;
  const int nc0 = n0 + nA, nc1 = n1 + nA;

  // ---- layer 1: X[16xKTOT] @ W1[KTOTx128] ----
  v8f acc0 = {}, acc1 = {};
#pragma unroll
  for (int k0 = 0; k0 < KTOT; k0 += 32) {
    Frag A, B0, B1;
    const unsigned short* ap = &Abuf[rowA * KTOT + k0 + kh];
    A.q[0] = *(const uint4*)ap;
    A.q[1] = *(const uint4*)(ap + 16);
    const unsigned short* b0p = W1T + (size_t)nc0 * KTOT + k0 + kh;
    B0.q[0] = *(const uint4*)b0p;
    B0.q[1] = *(const uint4*)(b0p + 16);
    const unsigned short* b1p = W1T + (size_t)nc1 * KTOT + k0 + kh;
    B1.q[0] = *(const uint4*)b1p;
    B1.q[1] = *(const uint4*)(b1p + 16);
    acc0 = __builtin_amdgcn_wmma_f32_16x16x32_bf16(false, A.v, false, B0.v,
                                                   (short)0, acc0, false, false);
    acc1 = __builtin_amdgcn_wmma_f32_16x16x32_bf16(false, A.v, false, B1.v,
                                                   (short)0, acc1, false, false);
  }
  {
    float bb0 = b1[nc0], bb1 = b1[nc1];
#pragma unroll
    for (int r = 0; r < 8; ++r) {
      int m = mBase + r;
      float x0 = acc0[r] + bb0;
      x0 = x0 / (1.f + __expf(-x0));       // SiLU
      float x1 = acc1[r] + bb1;
      x1 = x1 / (1.f + __expf(-x1));
      Hbuf[m * HDIM + nc0] = f2bf(x0);
      Hbuf[m * HDIM + nc1] = f2bf(x1);
    }
  }
  __syncthreads();

  // ---- layer 2: H[16x128] @ W2[128x128] ----
  v8f c0 = {}, c1 = {};
#pragma unroll
  for (int k0 = 0; k0 < HDIM; k0 += 32) {
    Frag A, B0, B1;
    const unsigned short* ap = &Hbuf[rowA * HDIM + k0 + kh];
    A.q[0] = *(const uint4*)ap;
    A.q[1] = *(const uint4*)(ap + 16);
    const unsigned short* b0p = W2T + (size_t)nc0 * HDIM + k0 + kh;
    B0.q[0] = *(const uint4*)b0p;
    B0.q[1] = *(const uint4*)(b0p + 16);
    const unsigned short* b1p = W2T + (size_t)nc1 * HDIM + k0 + kh;
    B1.q[0] = *(const uint4*)b1p;
    B1.q[1] = *(const uint4*)(b1p + 16);
    c0 = __builtin_amdgcn_wmma_f32_16x16x32_bf16(false, A.v, false, B0.v,
                                                 (short)0, c0, false, false);
    c1 = __builtin_amdgcn_wmma_f32_16x16x32_bf16(false, A.v, false, B1.v,
                                                 (short)0, c1, false, false);
  }
  {
    float bb0 = PMAP ? (nc0 < 34 ? b2[nc0] : 0.f) : b2[nc0];
    float bb1 = PMAP ? (nc1 < 34 ? b2[nc1] : 0.f) : b2[nc1];
#pragma unroll
    for (int r = 0; r < 8; ++r) {
      int m = mBase + r;
      Obuf[m * HDIM + nc0] = c0[r] + bb0;
      Obuf[m * HDIM + nc1] = c1[r] + bb1;
    }
  }
  __syncthreads();

  // ---- epilogue: per-row LayerNorm (wave handles 4 rows) + stores ----
#pragma unroll
  for (int j = 0; j < 4; ++j) {
    int row = wv * 4 + j;
    int rg = tile0 + row;
    int n = lane * 4;
    float4 x = *(const float4*)&Obuf[row * HDIM + n];
    if (LN) {
      float s = x.x + x.y + x.z + x.w;
      float sq = x.x * x.x + x.y * x.y + x.z * x.z + x.w * x.w;
#pragma unroll
      for (int o = 16; o >= 1; o >>= 1) {
        s += __shfl_xor(s, o, 32);
        sq += __shfl_xor(sq, o, 32);
      }
      float mu = s * (1.f / HDIM);
      float var = sq * (1.f / HDIM) - mu * mu;
      float rstd = rsqrtf(var + 1e-5f);
      float4 g = *(const float4*)&lnS[n];
      float4 bb = *(const float4*)&lnB[n];
      x.x = (x.x - mu) * rstd * g.x + bb.x;
      x.y = (x.y - mu) * rstd * g.y + bb.y;
      x.z = (x.z - mu) * rstd * g.z + bb.z;
      x.w = (x.w - mu) * rstd * g.w + bb.w;
    }
    if (rg >= nrows) continue;
    if (outScatter) {                      // segment_sum of m (pre-residual)
      int d = sidx[rg];
      float* o = outScatter + (size_t)d * HDIM + n;
      atomicAdd(o + 0, x.x);
      atomicAdd(o + 1, x.y);
      atomicAdd(o + 2, x.z);
      atomicAdd(o + 3, x.w);
    }
    if (PMAP) {
      float vals[4] = {x.x, x.y, x.z, x.w};
#pragma unroll
      for (int e = 0; e < 4; ++e) {
        int nn = n + e;
        if (nn < 34) {
          float v = vals[e];
          if (nn >= 17) v = (v > 15.f) ? v : log1pf(__expf(v));  // softplus
          outRow[(size_t)rg * 34 + nn] = v;
        }
      }
    } else if (outRow) {
      if (resid) {
        float4 rr = *(const float4*)(resid + (size_t)rg * HDIM + n);
        x.x += rr.x; x.y += rr.y; x.z += rr.z; x.w += rr.w;
      }
      *(float4*)(outRow + (size_t)rg * HDIM + n) = x;
    }
  }
}

// ---------------------------------------------------------------------------
// Host orchestration
// ---------------------------------------------------------------------------
namespace {

struct InetW {
  unsigned short *aw0, *aw1, *ew0, *ew1;
};

// params leaf bases (after 31 data arrays), jax sorted-dict flatten order:
// mlp(ln): W0,W1,b0,b1,ln_s,ln_b ; inet: aggr(6) then edge(6)
enum {
  P_DOWN0 = 31, P_DOWN1 = 43, P_G2M = 55, P_GRIDUP = 67,
  P_ID00 = 73, P_ID01 = 85, P_ID10 = 97, P_ID11 = 109,
  P_IU00 = 121, P_IU01 = 133, P_IU10 = 145, P_IU11 = 157,
  P_IU20 = 169, P_IU21 = 181,
  P_M2G = 193, P_PMAP = 205, P_UP0 = 209, P_UP1 = 221
};

}  // namespace

extern "C" void kernel_launch(void* const* d_in, const int* in_sizes, int n_in,
                              void* d_out, int out_size, void* d_ws,
                              size_t ws_size, hipStream_t stream) {
  (void)in_sizes; (void)n_in; (void)out_size; (void)ws_size;

  const int NG = 65536;
  const int NM0 = 16384, NM1 = 4096, NM2 = 1024;
  const int EG2M = 131072, EM2G = 131072;
  const int EM2M0 = 65536, EM2M1 = 16384, EM2M2 = 4096;
  const int EUP0 = 32768, EUP1 = 8192;
  const int EDN0 = 32768, EDN1 = 8192;

  auto Fp = [&](int i) { return (const float*)d_in[i]; };
  auto Ip = [&](int i) { return (const int*)d_in[i]; };

  const float* grid_rep = Fp(0);
  const float* mesh_in[3] = {Fp(1), Fp(2), Fp(3)};
  const float* g2m_emb = Fp(4);
  const float* m2g_emb = Fp(5);
  const float* m2m_emb[3] = {Fp(6), Fp(7), Fp(8)};
  const float* up_emb[2] = {Fp(9), Fp(10)};
  const float* dn_emb[2] = {Fp(11), Fp(12)};
  const int* g2m_src = Ip(13); const int* g2m_dst = Ip(14);
  const int* m2g_src = Ip(15); const int* m2g_dst = Ip(16);
  const int* m2m_src[3] = {Ip(17), Ip(19), Ip(21)};
  const int* m2m_dst[3] = {Ip(18), Ip(20), Ip(22)};
  const int* up_src[2] = {Ip(23), Ip(25)};
  const int* up_dst[2] = {Ip(24), Ip(26)};
  const int* dn_src[2] = {Ip(27), Ip(29)};
  const int* dn_dst[2] = {Ip(28), Ip(30)};

  // ---- workspace bump allocator ----
  size_t off = 0;
  auto balloc = [&](size_t bytes) -> void* {
    void* p = (char*)d_ws + off;
    off += (bytes + 255) & ~(size_t)255;
    return p;
  };

  // ---- convert weights to transposed bf16 (per launch; deterministic) ----
  auto cvt = [&](int leaf, int K, int N, int Npad) -> unsigned short* {
    unsigned short* dst = (unsigned short*)balloc((size_t)Npad * K * 2);
    int total = Npad * K;
    cvt_wT_kernel<<<dim3((total + 255) / 256), dim3(256), 0, stream>>>(
        Fp(leaf), dst, K, N, Npad);
    return dst;
  };
  auto cvt_inet = [&](int B) -> InetW {
    InetW w;
    w.aw0 = cvt(B + 0, 256, 128, 128);
    w.aw1 = cvt(B + 1, 128, 128, 128);
    w.ew0 = cvt(B + 6, 384, 128, 128);
    w.ew1 = cvt(B + 7, 128, 128, 128);
    return w;
  };

  unsigned short* guW0 = cvt(P_GRIDUP + 0, 128, 128, 128);
  unsigned short* guW1 = cvt(P_GRIDUP + 1, 128, 128, 128);
  unsigned short* pmW0 = cvt(P_PMAP + 0, 128, 128, 128);
  unsigned short* pmW1 = cvt(P_PMAP + 1, 128, 34, 128);   // zero-padded cols
  InetW w_g2m = cvt_inet(P_G2M);
  InetW w_m2g = cvt_inet(P_M2G);
  InetW w_up[2] = {cvt_inet(P_UP0), cvt_inet(P_UP1)};
  InetW w_dn[2] = {cvt_inet(P_DOWN0), cvt_inet(P_DOWN1)};
  InetW w_iu[3][2] = {{cvt_inet(P_IU00), cvt_inet(P_IU01)},
                      {cvt_inet(P_IU10), cvt_inet(P_IU11)},
                      {cvt_inet(P_IU20), cvt_inet(P_IU21)}};
  InetW w_id[2][2] = {{cvt_inet(P_ID00), cvt_inet(P_ID01)},
                      {cvt_inet(P_ID10), cvt_inet(P_ID11)}};

  // ---- activation buffers ----
  float* residual = (float*)balloc((size_t)NG * HDIM * 4);
  float* decoded  = (float*)balloc((size_t)NG * HDIM * 4);
  float* aggrG    = (float*)balloc((size_t)NG * HDIM * 4);
  float* aggrM    = (float*)balloc((size_t)NM0 * HDIM * 4);
  float* cur0     = (float*)balloc((size_t)NM0 * HDIM * 4);
  float* cur1     = (float*)balloc((size_t)NM1 * HDIM * 4);
  float* cur2     = (float*)balloc((size_t)NM2 * HDIM * 4);
  float* e0       = (float*)balloc((size_t)EM2M0 * HDIM * 4);
  float* e1       = (float*)balloc((size_t)EM2M1 * HDIM * 4);
  float* e2       = (float*)balloc((size_t)EM2M2 * HDIM * 4);

  // ---- launch helpers ----
  auto run_edge = [&](const InetW& w, int B, const float* edge_in,
                      const float* send, const float* rec, const int* src,
                      const int* dst, int nE, int nRec, float* aggr,
                      float* edge_out) {
    hipMemsetAsync(aggr, 0, (size_t)nRec * HDIM * sizeof(float), stream);
    fused_mlp_kernel<3, true, false>
        <<<dim3((nE + 15) / 16), dim3(128), 0, stream>>>(
            edge_in, nullptr, send, src, rec, dst,
            w.ew0, Fp(B + 8), w.ew1, Fp(B + 9), Fp(B + 10), Fp(B + 11),
            edge_out ? edge_in : nullptr, edge_out, aggr, dst, nE);
  };
  auto run_node = [&](const InetW& w, int B, const float* node,
                      const float* aggr, const float* res, float* out, int n) {
    fused_mlp_kernel<2, true, false>
        <<<dim3((n + 15) / 16), dim3(128), 0, stream>>>(
            node, nullptr, aggr, nullptr, nullptr, nullptr,
            w.aw0, Fp(B + 2), w.aw1, Fp(B + 3), Fp(B + 4), Fp(B + 5),
            res, out, nullptr, nullptr, n);
  };
  // PropagationNet: out = aggr + aggrMLP([rec, aggr])
  auto run_prop = [&](const InetW& w, int B, const float* send,
                      const float* rec, const float* edge, const int* src,
                      const int* dst, int nE, int nRec, float* aggr,
                      float* out) {
    run_edge(w, B, edge, send, rec, src, dst, nE, nRec, aggr, nullptr);
    run_node(w, B, rec, aggr, aggr, out, nRec);
  };
  // InteractionNet (in place): node += aggrMLP([node,aggr]); edge_out = edge_in + m
  auto run_inet = [&](const InetW& w, int B, float* node, const float* edge_in,
                      float* edge_out, const int* src, const int* dst, int nE,
                      int nN, float* aggr) {
    run_edge(w, B, edge_in, node, node, src, dst, nE, nN, aggr, edge_out);
    run_node(w, B, node, aggr, node, node, nN);
  };

  // ================== model ==================
  // residual = grid_rep + grid_update(grid_rep)
  fused_mlp_kernel<1, true, false>
      <<<dim3(NG / 16), dim3(128), 0, stream>>>(
          grid_rep, nullptr, nullptr, nullptr, nullptr, nullptr,
          guW0, Fp(P_GRIDUP + 2), guW1, Fp(P_GRIDUP + 3),
          Fp(P_GRIDUP + 4), Fp(P_GRIDUP + 5),
          grid_rep, residual, nullptr, nullptr, NG);

  // encode: grid -> mesh0
  run_prop(w_g2m, P_G2M, grid_rep, mesh_in[0], g2m_emb, g2m_src, g2m_dst,
           EG2M, NM0, aggrM, cur0);

  // up path
  run_inet(w_iu[0][0], P_IU00, cur0, m2m_emb[0], e0, m2m_src[0], m2m_dst[0],
           EM2M0, NM0, aggrM);
  run_inet(w_iu[0][1], P_IU01, cur0, e0, e0, m2m_src[0], m2m_dst[0],
           EM2M0, NM0, aggrM);                       // cur0 = mesh_reps[0], e0 = m2m_reps[0]
  run_prop(w_up[0], P_UP0, cur0, mesh_in[1], up_emb[0], up_src[0], up_dst[0],
           EUP0, NM1, aggrM, cur1);

  run_inet(w_iu[1][0], P_IU10, cur1, m2m_emb[1], e1, m2m_src[1], m2m_dst[1],
           EM2M1, NM1, aggrM);
  run_inet(w_iu[1][1], P_IU11, cur1, e1, e1, m2m_src[1], m2m_dst[1],
           EM2M1, NM1, aggrM);                       // cur1 = mesh_reps[1], e1 = m2m_reps[1]
  run_prop(w_up[1], P_UP1, cur1, mesh_in[2], up_emb[1], up_src[1], up_dst[1],
           EUP1, NM2, aggrM, cur2);

  run_inet(w_iu[2][0], P_IU20, cur2, m2m_emb[2], e2, m2m_src[2], m2m_dst[2],
           EM2M2, NM2, aggrM);
  run_inet(w_iu[2][1], P_IU21, cur2, e2, e2, m2m_src[2], m2m_dst[2],
           EM2M2, NM2, aggrM);

  // down path: l = 1
  run_prop(w_dn[1], P_DOWN1, cur2, cur1, dn_emb[1], dn_src[1], dn_dst[1],
           EDN1, NM1, aggrM, cur1);
  run_inet(w_id[1][0], P_ID10, cur1, e1, e1, m2m_src[1], m2m_dst[1],
           EM2M1, NM1, aggrM);
  run_inet(w_id[1][1], P_ID11, cur1, e1, e1, m2m_src[1], m2m_dst[1],
           EM2M1, NM1, aggrM);
  // l = 0
  run_prop(w_dn[0], P_DOWN0, cur1, cur0, dn_emb[0], dn_src[0], dn_dst[0],
           EDN0, NM0, aggrM, cur0);
  run_inet(w_id[0][0], P_ID00, cur0, e0, e0, m2m_src[0], m2m_dst[0],
           EM2M0, NM0, aggrM);
  run_inet(w_id[0][1], P_ID01, cur0, e0, e0, m2m_src[0], m2m_dst[0],
           EM2M0, NM0, aggrM);

  // decode: mesh0 -> grid
  run_prop(w_m2g, P_M2G, cur0, residual, m2g_emb, m2g_src, m2g_dst,
           EM2G, NG, aggrG, decoded);

  // param_map: [mean | softplus(raw)] -> d_out [NG, 34]
  fused_mlp_kernel<1, false, true>
      <<<dim3(NG / 16), dim3(128), 0, stream>>>(
          decoded, nullptr, nullptr, nullptr, nullptr, nullptr,
          pmW0, Fp(P_PMAP + 2), pmW1, Fp(P_PMAP + 3), nullptr, nullptr,
          nullptr, (float*)d_out, nullptr, nullptr, NG);
}
